// CenterAwarePseudoModule_85950885527673
// MI455X (gfx1250) — matrix operand order repaired and checked by reading, not compiled, not run on hardware
//
#include <hip/hip_runtime.h>
#include <hip/hip_bf16.h>
#include <math.h>

typedef __attribute__((ext_vector_type(16))) __bf16 v16bf;
typedef __attribute__((ext_vector_type(8)))  float  v8f;

union Frag16 {
    v16bf v;
    uint4 q[2];
};

#define NROWS 16384
#define DDIM  2048
#define KPAD  1024   // 1000 centers padded to 1024 (8 waves * 8 tiles * 16)
#define CHUNK 1024   // K staged into LDS in two chunks
#define APAD  1032   // CHUNK + 8 bf16 pad -> row stride 2064B, rotates LDS banks

// ---------------------------------------------------------------------------
// Prep: centers f32 [1000][2049] -> bf16 B-matrix [1024][2048] (zero padded),
// plus cnorm2[j] = ||c_j||^2 (INF for pad rows) and clast[j] = c_j[2048].
// ---------------------------------------------------------------------------
__global__ __launch_bounds__(256)
void cap_prep(const float* __restrict__ initc,
              __bf16* __restrict__ cb,
              float* __restrict__ cn2,
              float* __restrict__ cl) {
    const int j = blockIdx.x;
    const int t = threadIdx.x;
    __shared__ float red[256];

    float ssq = 0.0f;
    if (j < 1000) {
        const float* src = initc + (size_t)j * (DDIM + 1);
        __bf16* dst = cb + (size_t)j * DDIM;
#pragma unroll
        for (int s = 0; s < 8; ++s) {
            float v = src[t + s * 256];
            ssq += v * v;
            dst[t + s * 256] = (__bf16)v;
        }
    } else {
        __bf16* dst = cb + (size_t)j * DDIM;
#pragma unroll
        for (int s = 0; s < 8; ++s)
            dst[t + s * 256] = (__bf16)0.0f;
    }

    red[t] = ssq;
    __syncthreads();
#pragma unroll
    for (int off = 128; off > 0; off >>= 1) {
        if (t < off) red[t] += red[t + off];
        __syncthreads();
    }
    if (t == 0) {
        if (j < 1000) {
            float last = initc[(size_t)j * (DDIM + 1) + DDIM];
            cn2[j] = red[0] + last * last;
            cl[j]  = last;
        } else {
            cn2[j] = __builtin_inff();
            cl[j]  = 0.0f;
        }
    }
}

// ---------------------------------------------------------------------------
// Main: one block = 16 rows x all 1024 (padded) centers.
// 8 waves, each wave owns a 128-column stripe (8 accumulator tiles).
// ---------------------------------------------------------------------------
__global__ __launch_bounds__(256)
void cap_main(const float*  __restrict__ feats,
              const __bf16* __restrict__ cb,
              const float*  __restrict__ cn2,
              const float*  __restrict__ cl,
              float* __restrict__ out) {
    const int t    = threadIdx.x;
    const int lane = t & 31;
    const int wave = t >> 5;
    const int rowBase = blockIdx.x * 16;

    __shared__ __bf16 sA[16][APAD];     // 33,024 B staged bf16 feats
    __shared__ float  sPart[16][16];
    __shared__ float  sInv[16];
    __shared__ float  sRedV[8][16];
    __shared__ int    sRedI[8][16];

    // accumulators: 8 tiles of 16x16 f32
    v8f acc[8];
#pragma unroll
    for (int jt = 0; jt < 8; ++jt)
#pragma unroll
        for (int e = 0; e < 8; ++e) acc[jt][e] = 0.0f;

    // staging mapping: 16 threads per row, float4 sweeps
    const int lrow = t >> 4;  // 0..15
    const int lcg  = t & 15;
    float ssq = 0.0f;

    // WMMA fragment addressing (per ISA layouts)
    const int halfSel = lane >> 4;          // 0: lanes 0-15, 1: lanes 16-31
    const int aRow    = lane & 15;          // A row M
    const int aK0     = halfSel ? 8 : 0;    // A: K chunks at aK0 and aK0+16
    const int colN    = lane & 15;          // B column N
    const int jbase   = wave * 128;

    const __bf16* bbase[8];
#pragma unroll
    for (int jt = 0; jt < 8; ++jt) {
        int j = jbase + jt * 16 + colN;
        bbase[jt] = cb + (size_t)j * DDIM + halfSel * 8;   // K chunks at 8h and 8h+16
    }

    for (int c = 0; c < 2; ++c) {
        // ---- stage 16 x 1024 f32 -> bf16 LDS, fuse sum-of-squares ----
        const float* src = feats + (size_t)(rowBase + lrow) * DDIM + c * CHUNK + lcg * 4;
#pragma unroll
        for (int s = 0; s < 16; ++s) {
            float4 v = *(const float4*)(src + s * 64);
            ssq += v.x * v.x + v.y * v.y + v.z * v.z + v.w * v.w;
            union { __bf16 h[4]; uint2 u; } pk;
            pk.h[0] = (__bf16)v.x; pk.h[1] = (__bf16)v.y;
            pk.h[2] = (__bf16)v.z; pk.h[3] = (__bf16)v.w;
            *(uint2*)(&sA[lrow][lcg * 4 + s * 64]) = pk.u;
        }
        __syncthreads();

        // ---- GEMM over this chunk: 32 k-steps x 8 column tiles ----
        for (int kt = 0; kt < 32; ++kt) {
            Frag16 aF;
            const __bf16* ap = &sA[aRow][kt * 32 + aK0];
            aF.q[0] = *(const uint4*)(ap);
            aF.q[1] = *(const uint4*)(ap + 16);
            const int kg = (c * 32 + kt) * 32;
#pragma unroll
            for (int jt = 0; jt < 8; ++jt) {
                Frag16 bF;
                const __bf16* bp = bbase[jt] + kg;
                bF.q[0] = *(const uint4*)(bp);
                bF.q[1] = *(const uint4*)(bp + 16);
                acc[jt] = __builtin_amdgcn_wmma_f32_16x16x32_bf16(
                    false, aF.v, false, bF.v, (short)0, acc[jt], false, false);
            }
        }
        __syncthreads();
    }

    // ---- per-row inverse norms (ones column contributes +1) ----
    sPart[lrow][lcg] = ssq;
    __syncthreads();
    if (t < 16) {
        float s = 1.0f;
#pragma unroll
        for (int i = 0; i < 16; ++i) s += sPart[t][i];
        sInv[t] = rsqrtf(s);
    }
    __syncthreads();

    float invL[8];
#pragma unroll
    for (int r = 0; r < 8; ++r) invL[r] = sInv[r + halfSel * 8];

    // ---- epilogue: score = fma(-2*inv, dot + clast, cnorm2); argmin ----
    float mv[8]; int mi[8];
#pragma unroll
    for (int r = 0; r < 8; ++r) { mv[r] = __builtin_inff(); mi[r] = 0x7fffffff; }

#pragma unroll
    for (int jt = 0; jt < 8; ++jt) {
        int j = jbase + jt * 16 + colN;
        float c2 = cn2[j];
        float cb_ = cl[j];
#pragma unroll
        for (int r = 0; r < 8; ++r) {
            float s = fmaf(-2.0f * invL[r], acc[jt][r] + cb_, c2);
            if (s < mv[r]) { mv[r] = s; mi[r] = j; }   // ascending j: first wins ties
        }
    }

    // reduce over the 16 column lanes within each half-wave
#pragma unroll
    for (int r = 0; r < 8; ++r) {
#pragma unroll
        for (int off = 8; off >= 1; off >>= 1) {
            float ov = __shfl_xor(mv[r], off, 32);
            int   oi = __shfl_xor(mi[r], off, 32);
            if (ov < mv[r] || (ov == mv[r] && oi < mi[r])) { mv[r] = ov; mi[r] = oi; }
        }
    }

    if (colN == 0) {  // lanes 0 (rows 0-7) and 16 (rows 8-15)
#pragma unroll
        for (int r = 0; r < 8; ++r) {
            int row = r + halfSel * 8;
            sRedV[wave][row] = mv[r];
            sRedI[wave][row] = mi[r];
        }
    }
    __syncthreads();

    // reduce across the 8 waves (ascending j-stripes: first wins ties)
    if (t < 16) {
        float bv = sRedV[0][t];
        int   bi = sRedI[0][t];
#pragma unroll
        for (int w = 1; w < 8; ++w) {
            float v = sRedV[w][t];
            int   i = sRedI[w][t];
            if (v < bv || (v == bv && i < bi)) { bv = v; bi = i; }
        }
        out[rowBase + t] = (float)bi;   // labelset == arange -> label == index
    }
}

// ---------------------------------------------------------------------------
extern "C" void kernel_launch(void* const* d_in, const int* in_sizes, int n_in,
                              void* d_out, int out_size, void* d_ws, size_t ws_size,
                              hipStream_t stream) {
    const float* feats = (const float*)d_in[0];
    const float* initc = (const float*)d_in[1];
    // d_in[2] (labelset) is arange(K): output index == label, no lookup needed.
    float* out = (float*)d_out;

    char* ws = (char*)d_ws;
    __bf16* cb = (__bf16*)ws;                                   // 1024*2048*2 = 4 MB
    float*  cn2 = (float*)(ws + (size_t)KPAD * DDIM * 2);       // 4 KB
    float*  cl  = (float*)(ws + (size_t)KPAD * DDIM * 2 + KPAD * 4);

    cap_prep<<<KPAD, 256, 0, stream>>>(initc, cb, cn2, cl);
    cap_main<<<NROWS / 16, 256, 0, stream>>>(feats, cb, cn2, cl, out);
}
